// GCNEncoderNoise_43688407335390
// MI455X (gfx1250) — matrix operand order repaired and checked by reading, not compiled, not run on hardware
//
#include <hip/hip_runtime.h>
#include <hip/hip_bf16.h>
#include <math.h>

// ---------------------------------------------------------------------------
// ChebConv (K=3) x 3 layers for MI455X (gfx1250, wave32, WMMA).
// GEMMs run on v_wmma_f32_16x16x32_bf16 (f32 accumulate) over bf16-staged
// operands; graph propagation (segment scatter-add) runs in fp32 atomics
// (all activations are L2-resident: max 41 MB << 192 MB L2).
// ---------------------------------------------------------------------------

typedef __attribute__((ext_vector_type(16))) __bf16 v16bf;
typedef __attribute__((ext_vector_type(8)))  __bf16 v8bf;
typedef __attribute__((ext_vector_type(8)))  float  v8f;

#define N_NODES 20000
#define N_EDGES 320000

// ----------------------------- utility hashes ------------------------------
static __device__ __forceinline__ uint32_t pcg_hash(uint32_t v) {
    uint32_t s = v * 747796405u + 2891336453u;
    uint32_t w = ((s >> ((s >> 28u) + 4u)) ^ s) * 277803737u;
    return (w >> 22u) ^ w;
}

// Deterministic standard-normal noise (Box-Muller over PCG hashes).
static __device__ __forceinline__ float gauss_noise(uint32_t layer, uint32_t idx) {
    uint32_t s  = idx * 0x9E3779B9u + layer * 0x7F4A7C15u;
    uint32_t h1 = pcg_hash(s);
    uint32_t h2 = pcg_hash(s ^ 0x85EBCA6Bu);
    float u1 = ((h1 >> 8) + 1u) * (1.0f / 16777216.0f);
    float u2 = (h2 >> 8) * (1.0f / 16777216.0f);
    return sqrtf(-2.0f * __logf(u1)) * __cosf(6.28318530718f * u2);
}

// ------------------------------ small kernels ------------------------------
__global__ void k_zero(float* __restrict__ p, size_t n) {
    size_t i = (size_t)blockIdx.x * blockDim.x + threadIdx.x;
    if (i < n) p[i] = 0.0f;
}

__global__ void k_deg(const int* __restrict__ row, const int* __restrict__ col,
                      float* __restrict__ deg, int E) {
    int e = blockIdx.x * blockDim.x + threadIdx.x;
    if (e < E && row[e] != col[e]) atomicAdd(&deg[row[e]], 1.0f);
}

__global__ void k_dinv(float* __restrict__ deg, int n) {
    int i = blockIdx.x * blockDim.x + threadIdx.x;
    if (i < n) {
        float d = deg[i];
        deg[i] = (d > 0.0f) ? rsqrtf(fmaxf(d, 1e-12f)) : 0.0f;
    }
}

__global__ void k_normw(const int* __restrict__ row, const int* __restrict__ col,
                        const float* __restrict__ dinv, float* __restrict__ nw, int E) {
    int e = blockIdx.x * blockDim.x + threadIdx.x;
    if (e < E) {
        int r = row[e], c = col[e];
        nw[e] = (r != c) ? (-dinv[r] * dinv[c]) : 0.0f;
    }
}

// Pad N x 86 input into N x 96 (zero pad -> pads never contribute to GEMM/prop).
__global__ void k_pad_input(const float* __restrict__ v, float* __restrict__ tx0, size_t n) {
    size_t i = (size_t)blockIdx.x * blockDim.x + threadIdx.x;
    if (i < n) {
        size_t node = i / 96, f = i % 96;
        tx0[i] = (f < 86) ? v[node * 86 + f] : 0.0f;
    }
}

// Streaming fp32 -> bf16 conversion, 4 elements / thread (packed cvt + b64 store).
__global__ void k_cvt_bf16(const float* __restrict__ in, __bf16* __restrict__ out, size_t n4) {
    size_t i = (size_t)blockIdx.x * blockDim.x + threadIdx.x;
    if (i < n4) {
        float4 f = ((const float4*)in)[i];
        __bf16 b[4] = { (__bf16)f.x, (__bf16)f.y, (__bf16)f.z, (__bf16)f.w };
        ((ulonglong1*)out)[i] = *(const ulonglong1*)b;   // 8-byte store
    }
}

// Convert + transpose weights: W[k][ci][co] (f32) -> Wt[k][co][ci_pad] (bf16).
__global__ void k_wcvt(const float* __restrict__ W, __bf16* __restrict__ Wt,
                       int Cin, int Cinp, int Cout) {
    size_t total = (size_t)3 * Cout * Cinp;
    size_t i = (size_t)blockIdx.x * blockDim.x + threadIdx.x;
    if (i < total) {
        int k  = (int)(i / ((size_t)Cout * Cinp));
        int r  = (int)(i % ((size_t)Cout * Cinp));
        int co = r / Cinp;
        int ci = r % Cinp;
        float w = (ci < Cin) ? W[((size_t)k * Cin + ci) * Cout + co] : 0.0f;
        Wt[i] = (__bf16)w;
    }
}

// y[row[e]] += norm_w[e] * x[col[e]]  over C features (4 per thread).
__global__ void k_prop(const int* __restrict__ row, const int* __restrict__ col,
                       const float* __restrict__ nw,
                       const float* __restrict__ x, float* __restrict__ y,
                       int E, int C, int stride) {
    int nch = (C + 3) >> 2;
    size_t tid = (size_t)blockIdx.x * blockDim.x + threadIdx.x;
    int e = (int)(tid / nch);
    int c0 = (int)(tid % nch) * 4;
    if (e >= E) return;
    float w = nw[e];
    if (w == 0.0f) return;
    const float* xp = x + (size_t)col[e] * stride + c0;   // in-bounds: stride >= c0+4
    float*       yp = y + (size_t)row[e] * stride + c0;
#pragma unroll
    for (int j = 0; j < 4; ++j) {
        if (c0 + j < C) atomicAdd(&yp[j], w * xp[j]);
    }
}

// txB = 2*txB - tx0 (Chebyshev recurrence T2 = 2*L*T1 - T0), over padded region.
__global__ void k_cheb2(float* __restrict__ txB, const float* __restrict__ tx0, size_t n) {
    size_t i = (size_t)blockIdx.x * blockDim.x + threadIdx.x;
    if (i < n) txB[i] = 2.0f * txB[i] - tx0[i];
}

// --------------------------- fused WMMA GEMM -------------------------------
// out[m, n] = noise * relu( bias[n] + sum_t sum_k A_t[m,k] * Wt[t][n][k] )
// One wave computes a 16x32 tile: one shared A fragment feeds two WMMAs.
// A fragment, same contiguous k order as B (bf16-staged, two b128 loads):
//   lane l (M = l&15): halves 0..7  hold K = kk*32 + 16*(l>>4) + 0..7
//                      halves 8..15 hold K = kk*32 + 16*(l>>4) + 8..15
// B fragment in this code's contiguous k order (Wt transposed so K contiguous):
//   lane l (N = l&15): halves 0..15 hold K = kk*32 + 16*(l>>4) + 0..15
__global__ __launch_bounds__(256)
void k_cheb_gemm(const __bf16* __restrict__ A0, const __bf16* __restrict__ A1,
                 const __bf16* __restrict__ A2,
                 const __bf16* __restrict__ Wt, const float* __restrict__ bias,
                 float* __restrict__ out, int n, int Cinp, int Cout, uint32_t layer) {
    const int lane  = threadIdx.x & 31;
    const int wave  = threadIdx.x >> 5;
    const int tileM = blockIdx.x * 128 + wave * 16;
    const int tileN = blockIdx.y * 32;
    const int hi    = lane >> 4;
    const int mrow  = tileM + (lane & 15);
    const int m_ld  = (mrow < n) ? mrow : (n - 1);   // clamp; stores are masked
    const int nc0   = tileN + (lane & 15);
    const int nc1   = nc0 + 16;
    const int koffA = hi * 16;
    const int ksteps = Cinp >> 5;

    v8f acc0 = {}, acc1 = {};
    for (int t = 0; t < 3; ++t) {
        const __bf16* arow  = ((t == 0) ? A0 : (t == 1) ? A1 : A2)
                              + (size_t)m_ld * Cinp + koffA;
        const __bf16* bbase = Wt + (size_t)t * Cout * Cinp + hi * 16;
        const __bf16* brow0 = bbase + (size_t)nc0 * Cinp;
        const __bf16* brow1 = bbase + (size_t)nc1 * Cinp;
        for (int kk = 0; kk < ksteps; ++kk) {
            v8bf lo = *(const v8bf*)(arow + kk * 32);        // K = koff .. koff+7
            v8bf hi8 = *(const v8bf*)(arow + kk * 32 + 8);   // K = koff+8 .. koff+15
            v16bf a;
#pragma unroll
            for (int j = 0; j < 8; ++j) { a[j] = lo[j]; a[8 + j] = hi8[j]; }
            v16bf b0 = *(const v16bf*)(brow0 + kk * 32);     // 32B contiguous
            v16bf b1 = *(const v16bf*)(brow1 + kk * 32);
            acc0 = __builtin_amdgcn_wmma_f32_16x16x32_bf16(
                       false, a, false, b0, (short)0, acc0, false, false);
            acc1 = __builtin_amdgcn_wmma_f32_16x16x32_bf16(
                       false, a, false, b1, (short)0, acc1, false, false);
        }
    }

    float bv0 = bias[nc0];
    float bv1 = bias[nc1];
#pragma unroll
    for (int r = 0; r < 8; ++r) {
        int m = tileM + r + hi * 8;   // C/D layout: vgpr r, lanes 16-31 -> M = r+8
        if (m < n) {
            float v0 = fmaxf(acc0[r] + bv0, 0.0f);
            float v1 = fmaxf(acc1[r] + bv1, 0.0f);
            v0 *= gauss_noise(layer, (uint32_t)(m * Cout + nc0));
            v1 *= gauss_noise(layer, (uint32_t)(m * Cout + nc1));
            out[(size_t)m * Cout + nc0] = v0;
            out[(size_t)m * Cout + nc1] = v1;
        }
    }
}

// ------------------------------- launcher ----------------------------------
static inline unsigned blocks_for(size_t n, unsigned t) { return (unsigned)((n + t - 1) / t); }

extern "C" void kernel_launch(void* const* d_in, const int* in_sizes, int n_in,
                              void* d_out, int out_size, void* d_ws, size_t ws_size,
                              hipStream_t stream) {
    const float* v     = (const float*)d_in[0];
    const int*   edges = (const int*)d_in[1];
    const float* W1    = (const float*)d_in[2];
    const float* b1    = (const float*)d_in[3];
    const float* W2    = (const float*)d_in[4];
    const float* b2    = (const float*)d_in[5];
    const float* W3    = (const float*)d_in[6];
    const float* b3    = (const float*)d_in[7];
    const int* row = edges;
    const int* col = edges + N_EDGES;

    // ---- workspace carve-out (256B aligned slabs) ----
    char* wsb = (char*)d_ws;
    size_t off = 0;
    auto carve = [&](size_t bytes) -> char* {
        char* p = wsb + off;
        off = (off + bytes + 255) & ~(size_t)255;
        return p;
    };
    float*  deg  = (float*)carve((size_t)N_NODES * 4);
    float*  nw   = (float*)carve((size_t)N_EDGES * 4);
    float*  tx0p = (float*)carve((size_t)N_NODES * 96 * 4);
    float*  txA  = (float*)carve((size_t)N_NODES * 256 * 4);
    float*  txB  = (float*)carve((size_t)N_NODES * 256 * 4);
    __bf16* a0b  = (__bf16*)carve((size_t)N_NODES * 256 * 2);   // bf16-staged Tx0
    __bf16* a1b  = (__bf16*)carve((size_t)N_NODES * 256 * 2);   // bf16-staged Tx1
    __bf16* a2b  = (__bf16*)carve((size_t)N_NODES * 256 * 2);   // bf16-staged Tx2
    __bf16* w1t  = (__bf16*)carve((size_t)3 * 128 * 96 * 2);
    __bf16* w2t  = (__bf16*)carve((size_t)3 * 256 * 128 * 2);
    __bf16* w3t  = (__bf16*)carve((size_t)3 * 512 * 256 * 2);

    float* x1 = (float*)d_out;                    // N x 128
    float* x2 = x1 + (size_t)N_NODES * 128;       // N x 256
    float* x3 = x2 + (size_t)N_NODES * 256;       // N x 512

    // ---- graph normalization: deg -> dinv -> norm_w ----
    k_zero<<<blocks_for(N_NODES, 256), 256, 0, stream>>>(deg, N_NODES);
    k_deg<<<blocks_for(N_EDGES, 256), 256, 0, stream>>>(row, col, deg, N_EDGES);
    k_dinv<<<blocks_for(N_NODES, 256), 256, 0, stream>>>(deg, N_NODES);
    k_normw<<<blocks_for(N_EDGES, 256), 256, 0, stream>>>(row, col, deg, nw, N_EDGES);

    // ---- input padding + weight conversion ----
    k_pad_input<<<blocks_for((size_t)N_NODES * 96, 256), 256, 0, stream>>>(
        v, tx0p, (size_t)N_NODES * 96);
    k_wcvt<<<blocks_for((size_t)3 * 128 * 96, 256), 256, 0, stream>>>(W1, w1t, 86, 96, 128);
    k_wcvt<<<blocks_for((size_t)3 * 256 * 128, 256), 256, 0, stream>>>(W2, w2t, 128, 128, 256);
    k_wcvt<<<blocks_for((size_t)3 * 512 * 256, 256), 256, 0, stream>>>(W3, w3t, 256, 256, 512);

    // ---- one ChebConv layer ----
    auto run_layer = [&](const float* tx0, int Cin, int Cinp, int Cout,
                         const __bf16* wt, const float* bias, float* outp, uint32_t layer) {
        size_t nel = (size_t)N_NODES * Cinp;
        size_t n4  = nel >> 2;                 // Cinp is a multiple of 4
        int nch = (Cin + 3) >> 2;
        size_t nprop = (size_t)N_EDGES * nch;

        k_zero<<<blocks_for(nel, 256), 256, 0, stream>>>(txA, nel);
        k_prop<<<blocks_for(nprop, 256), 256, 0, stream>>>(row, col, nw, tx0, txA,
                                                           N_EDGES, Cin, Cinp);
        k_zero<<<blocks_for(nel, 256), 256, 0, stream>>>(txB, nel);
        k_prop<<<blocks_for(nprop, 256), 256, 0, stream>>>(row, col, nw, txA, txB,
                                                           N_EDGES, Cin, Cinp);
        k_cheb2<<<blocks_for(nel, 256), 256, 0, stream>>>(txB, tx0, nel);

        // Stage all three Chebyshev operands to bf16 once (amortized over
        // Cout/32 column tiles that re-read them in the GEMM).
        k_cvt_bf16<<<blocks_for(n4, 256), 256, 0, stream>>>(tx0, a0b, n4);
        k_cvt_bf16<<<blocks_for(n4, 256), 256, 0, stream>>>(txA, a1b, n4);
        k_cvt_bf16<<<blocks_for(n4, 256), 256, 0, stream>>>(txB, a2b, n4);

        dim3 grid(blocks_for(N_NODES, 128), Cout / 32);
        k_cheb_gemm<<<grid, 256, 0, stream>>>(a0b, a1b, a2b, wt, bias, outp,
                                              N_NODES, Cinp, Cout, layer);
    };

    run_layer(tx0p, 86, 96, 128, w1t, b1, x1, 1u);   // x1 feeds layer 2
    run_layer(x1, 128, 128, 256, w2t, b2, x2, 2u);   // x2 feeds layer 3
    run_layer(x2, 256, 256, 512, w3t, b3, x3, 3u);
}
